// CustomLlamaMLAForInfer_11690900979842
// MI455X (gfx1250) — matrix operand order repaired and checked by reading, not compiled, not run on hardware
//
#include <hip/hip_runtime.h>
#include <hip/hip_bf16.h>

typedef __bf16 bf16;
typedef __attribute__((ext_vector_type(16))) __bf16 v16bf;
typedef __attribute__((ext_vector_type(8)))  __bf16 v8bf;
typedef __attribute__((ext_vector_type(8)))  float  v8f;
typedef __attribute__((ext_vector_type(4)))  int    v4i;

__device__ __forceinline__ v8f wmma_bf16(v16bf a, v16bf b, v8f c) {
  return __builtin_amdgcn_wmma_f32_16x16x32_bf16(false, a, false, b, (short)0, c,
                                                 false, false);
}

union Frag16 { v16bf v; v8bf h[2]; };

// ---------------------------------------------------------------------------
// CDNA5 async global->LDS copy (ASYNCcnt-tracked, no VGPR round trip).
// Builtin takes (global v4i*, lds v4i*, imm offset, imm cpol). Pointers are
// built via integer casts: generic->as3 truncates to the 32-bit LDS aperture
// offset (ISA 10.2), generic->as1 is the identity.
// ---------------------------------------------------------------------------
#if defined(__gfx1250__) && __has_builtin(__builtin_amdgcn_global_load_async_to_lds_b128)
#define HAS_ASYNC_COPY 1
__device__ __forceinline__ void async_copy16(const void* g, void* l) {
  __builtin_amdgcn_global_load_async_to_lds_b128(
      (__attribute__((address_space(1))) v4i*)(unsigned long long)g,
      (__attribute__((address_space(3))) v4i*)(unsigned)(unsigned long long)l,
      0, 0);
}
__device__ __forceinline__ void async_wait0() {
#if __has_builtin(__builtin_amdgcn_s_wait_asynccnt)
  __builtin_amdgcn_s_wait_asynccnt(0);
#else
  asm volatile("s_wait_asynccnt 0x0" ::: "memory");
#endif
}
#endif

// ---------------------------------------------------------------------------
// f32 -> bf16 convert
// ---------------------------------------------------------------------------
__global__ __launch_bounds__(256) void cvt_f32_bf16(const float* __restrict__ in,
                                                    bf16* __restrict__ out, int n) {
  int i = blockIdx.x * 256 + threadIdx.x;
  if (i < n) out[i] = (bf16)in[i];
}

// ---------------------------------------------------------------------------
// Generic bf16 GEMM: C[M,N] (f32) = A[M,K] * B[N,K]^T  (both bf16, K-major)
// block tile 128x128, BK=32, 256 threads = 8 waves (4 m-waves x 2 n-waves),
// each wave computes 32x64 via 2x4 WMMA 16x16x32 tiles.
// ---------------------------------------------------------------------------
#define BM 128
#define BN 128
#define BK 32
#define LDT 40   // padded LDS row stride (halfs)

__global__ __launch_bounds__(256) void gemm_bf16(const bf16* __restrict__ A,
                                                 const bf16* __restrict__ B,
                                                 float* __restrict__ C,
                                                 int M, int N, int K) {
  __shared__ __align__(16) bf16 As[BM * LDT];
  __shared__ __align__(16) bf16 Bs[BN * LDT];
  const int tid  = threadIdx.x;
  const int wave = tid >> 5, lane = tid & 31;
  const int wm = wave >> 1, wn = wave & 1;   // 4 x 2 wave grid
  const int hl = lane >> 4, l16 = lane & 15;
  const int mBase = blockIdx.x * BM;
  const int nBase = blockIdx.y * BN;

  v8f acc[2][4] = {};

  for (int kt = 0; kt < K; kt += BK) {
    __syncthreads();
    // stage 128x32 A tile and 128x32 B tile: 512 16B-chunks each, 2 per thread
#pragma unroll
    for (int i = 0; i < 2; ++i) {
      const int cid = tid + 256 * i;
      const int r = cid >> 2, c = (cid & 3) << 3;
#ifdef HAS_ASYNC_COPY
      async_copy16(&A[(size_t)(mBase + r) * K + kt + c], &As[r * LDT + c]);
      async_copy16(&B[(size_t)(nBase + r) * K + kt + c], &Bs[r * LDT + c]);
#else
      *(v8bf*)&As[r * LDT + c] = *(const v8bf*)&A[(size_t)(mBase + r) * K + kt + c];
      *(v8bf*)&Bs[r * LDT + c] = *(const v8bf*)&B[(size_t)(nBase + r) * K + kt + c];
#endif
      if (kt + BK < K) {   // CDNA5 global_prefetch_b8 of next tiles
        __builtin_prefetch(&A[(size_t)(mBase + r) * K + kt + BK + c], 0, 1);
        __builtin_prefetch(&B[(size_t)(nBase + r) * K + kt + BK + c], 0, 1);
      }
    }
#ifdef HAS_ASYNC_COPY
    async_wait0();
#endif
    __syncthreads();
#pragma unroll
    for (int sm = 0; sm < 2; ++sm) {
      Frag16 af;
      const int row = wm * 32 + sm * 16 + l16;
      af.h[0] = *(const v8bf*)&As[row * LDT + 8 * hl];
      af.h[1] = *(const v8bf*)&As[row * LDT + 16 + 8 * hl];
#pragma unroll
      for (int sn = 0; sn < 4; ++sn) {
        Frag16 bfr;
        const int col = wn * 64 + sn * 16 + l16;
        bfr.h[0] = *(const v8bf*)&Bs[col * LDT + 8 * hl];
        bfr.h[1] = *(const v8bf*)&Bs[col * LDT + 16 + 8 * hl];
        acc[sm][sn] = wmma_bf16(af.v, bfr.v, acc[sm][sn]);
      }
    }
  }
#pragma unroll
  for (int sm = 0; sm < 2; ++sm)
#pragma unroll
    for (int sn = 0; sn < 4; ++sn)
#pragma unroll
      for (int v = 0; v < 8; ++v) {
        const int m = mBase + wm * 32 + sm * 16 + v + 8 * hl;
        const int n = nBase + wn * 64 + sn * 16 + l16;
        C[(size_t)m * N + n] = acc[sm][sn][v];
      }
}

// ---------------------------------------------------------------------------
// RoPE on q: q_raw f32 [B*S,4096] -> qb bf16 [B,32,S,128]
// ---------------------------------------------------------------------------
__global__ __launch_bounds__(256) void rope_q(const float* __restrict__ qr,
                                              bf16* __restrict__ qb) {
  const int idx = blockIdx.x * 256 + threadIdx.x;
  const int i = idx & 63;
  const int h = (idx >> 6) & 31;
  const int t = idx >> 11;       // token = b*2048 + s
  const int s = t & 2047;
  const int b = t >> 11;
  const float inv = __expf(-(float)i * (9.210340371976184f / 64.f)); // theta^(-i/64)
  const float ang = (float)s * inv;
  const float c = __cosf(ang), sn = __sinf(ang);
  const float* row = qr + (size_t)t * 4096 + h * 128;
  const float a  = row[i];
  const float bb = row[i + 64];
  bf16* o = qb + (((size_t)(b * 32 + h)) * 2048 + s) * 128;
  o[i]      = (bf16)(a * c - bb * sn);
  o[i + 64] = (bf16)(bb * c + a * sn);
}

// ---------------------------------------------------------------------------
// Assemble full K: rope part from k_r0 (rotated), nope part from k_c.
// kb bf16 [B,8,S,128]
// ---------------------------------------------------------------------------
__global__ __launch_bounds__(256) void build_k(const float* __restrict__ kr0,
                                               const float* __restrict__ kc,
                                               bf16* __restrict__ kb) {
  const int idx = blockIdx.x * 256 + threadIdx.x;
  const int j  = idx & 63;
  const int kh = (idx >> 6) & 7;
  const int t  = idx >> 9;
  const int s  = t & 2047;
  const int b  = t >> 11;
  bf16* o = kb + (((size_t)(b * 8 + kh)) * 2048 + s) * 128;
  if (j < 32) {   // rope pair (dims j, j+64), freq index = j
    const float a  = kr0[(size_t)t * 512 + kh * 64 + j];
    const float bb = kr0[(size_t)t * 512 + kh * 64 + 32 + j];
    const float inv = __expf(-(float)j * (9.210340371976184f / 64.f));
    const float ang = (float)s * inv;
    const float c = __cosf(ang), sn = __sinf(ang);
    o[j]      = (bf16)(a * c - bb * sn);
    o[j + 64] = (bf16)(bb * c + a * sn);
  } else {        // nope: dim j <- kc[kh*64+(j-32)]; dim j+64 <- kc[kh*64+j]
    o[j]      = (bf16)kc[(size_t)t * 512 + kh * 64 + (j - 32)];
    o[j + 64] = (bf16)kc[(size_t)t * 512 + kh * 64 + j];
  }
}

// ---------------------------------------------------------------------------
// V reshape: v_raw f32 [B*S,1024] -> vb bf16 [B,8,S,128]
// ---------------------------------------------------------------------------
__global__ __launch_bounds__(256) void reshape_v(const float* __restrict__ vr,
                                                 bf16* __restrict__ vb) {
  const int idx = blockIdx.x * 256 + threadIdx.x;
  const int d  = idx & 127;
  const int kh = (idx >> 7) & 7;
  const int t  = idx >> 10;
  const int s  = t & 2047;
  const int b  = t >> 11;
  vb[(((size_t)(b * 8 + kh)) * 2048 + s) * 128 + d] = (bf16)vr[(size_t)t * 1024 + kh * 128 + d];
}

// ---------------------------------------------------------------------------
// Causal flash attention (GQA 4:1), bf16 WMMA, online softmax.
// grid (S/64, B*32); block 128 (4 waves); wave w owns q rows [64*bx+16w, +16).
// ---------------------------------------------------------------------------
__global__ __launch_bounds__(128) void mla_attn(const bf16* __restrict__ qb,
                                                const bf16* __restrict__ kb,
                                                const bf16* __restrict__ vb,
                                                bf16* __restrict__ attn) {
  constexpr int S = 2048, D = 128;
  __shared__ __align__(16) bf16 Ks[64 * 136];     // K tile, row-major [key][d]
  __shared__ __align__(16) bf16 Vt[128 * 72];     // V tile, transposed [d][key]
  __shared__ __align__(16) bf16 Ps[4 * 16 * 64];  // per-wave P staging
  const int tid = threadIdx.x, wave = tid >> 5, lane = tid & 31;
  const int hl = lane >> 4, l16 = lane & 15;
  const int bx = blockIdx.x;
  const int bh = blockIdx.y;
  const int b = bh >> 5, h = bh & 31;
  const int kh = h >> 2;
  const bf16* Q  = qb + ((size_t)(b * 32 + h)) * S * D;
  const bf16* Kp = kb + ((size_t)(b * 8 + kh)) * S * D;
  const bf16* Vp = vb + ((size_t)(b * 8 + kh)) * S * D;
  const int qsb = bx * 64;
  const int qrow0 = qsb + wave * 16;

  // Q fragments (A-matrix layout), 4 chunks of K-dim 32 over d=128
  Frag16 qf[4];
#pragma unroll
  for (int kc = 0; kc < 4; ++kc) {
    const bf16* base = Q + (size_t)(qrow0 + l16) * D + kc * 32;
    qf[kc].h[0] = *(const v8bf*)(base + 8 * hl);
    qf[kc].h[1] = *(const v8bf*)(base + 16 + 8 * hl);
  }

  v8f acc[8] = {};
  float mrow[8], lrow[8];
#pragma unroll
  for (int v = 0; v < 8; ++v) { mrow[v] = -1e30f; lrow[v] = 0.f; }
  const float scale = 0.08838834764831845f;   // 1/sqrt(128)

  for (int kt = 0; kt <= bx; ++kt) {
    __syncthreads();
    // stage K tile (row-major, async DMA when available) and V tile (transposed)
#pragma unroll
    for (int i = 0; i < 8; ++i) {
      const int cid = tid + 128 * i;          // 1024 chunks of 8 halfs
      const int r = cid >> 4, c = (cid & 15) << 3;
#ifdef HAS_ASYNC_COPY
      async_copy16(&Kp[(size_t)(kt * 64 + r) * D + c], &Ks[r * 136 + c]);
#else
      *(v8bf*)&Ks[r * 136 + c] = *(const v8bf*)&Kp[(size_t)(kt * 64 + r) * D + c];
#endif
      v8bf tmp = *(const v8bf*)&Vp[(size_t)(kt * 64 + r) * D + c];
#pragma unroll
      for (int j = 0; j < 8; ++j) Vt[(c + j) * 72 + r] = tmp[j];
    }
#ifdef HAS_ASYNC_COPY
    async_wait0();
#endif
    __syncthreads();

    // scores S = Q K^T  (4 key subtiles of 16)
    v8f sc[4];
#pragma unroll
    for (int nb = 0; nb < 4; ++nb) {
      v8f s = {};
#pragma unroll
      for (int kc = 0; kc < 4; ++kc) {
        Frag16 kf;
        const int key = nb * 16 + l16;
        kf.h[0] = *(const v8bf*)&Ks[key * 136 + kc * 32 + 8 * hl];
        kf.h[1] = *(const v8bf*)&Ks[key * 136 + kc * 32 + 16 + 8 * hl];
        s = wmma_bf16(qf[kc].v, kf.v, s);
      }
      sc[nb] = s;
    }
    // scale + causal mask (only the diagonal tile can violate causality)
    const bool maskTile = (kt == bx);
#pragma unroll
    for (int nb = 0; nb < 4; ++nb)
#pragma unroll
      for (int v = 0; v < 8; ++v) {
        float x = sc[nb][v] * scale;
        if (maskTile) {
          const int key = kt * 64 + nb * 16 + l16;
          const int qr  = qrow0 + v + 8 * hl;
          if (key > qr) x = -1e9f;
        }
        sc[nb][v] = x;
      }
    // online softmax per row (row v lives across 16 lanes of each half)
#pragma unroll
    for (int v = 0; v < 8; ++v) {
      float tmax = fmaxf(fmaxf(sc[0][v], sc[1][v]), fmaxf(sc[2][v], sc[3][v]));
#pragma unroll
      for (int m = 1; m < 16; m <<= 1) tmax = fmaxf(tmax, __shfl_xor(tmax, m, 32));
      const float mnew  = fmaxf(mrow[v], tmax);
      const float alpha = __expf(mrow[v] - mnew);
      mrow[v] = mnew;
      float rs = 0.f;
#pragma unroll
      for (int nb = 0; nb < 4; ++nb) {
        const float p = __expf(sc[nb][v] - mnew);
        sc[nb][v] = p;
        rs += p;
      }
#pragma unroll
      for (int m = 1; m < 16; m <<= 1) rs += __shfl_xor(rs, m, 32);
      lrow[v] = lrow[v] * alpha + rs;
#pragma unroll
      for (int nb2 = 0; nb2 < 8; ++nb2) acc[nb2][v] *= alpha;
    }
    // P (C-layout) -> LDS -> reload as A-layout fragments (per-wave region)
    bf16* Pw = &Ps[wave * 16 * 64];
#pragma unroll
    for (int nb = 0; nb < 4; ++nb)
#pragma unroll
      for (int v = 0; v < 8; ++v)
        Pw[(v + 8 * hl) * 64 + nb * 16 + l16] = (bf16)sc[nb][v];
    asm volatile("s_wait_dscnt 0x0" ::: "memory");
    Frag16 pf[2];
#pragma unroll
    for (int ka = 0; ka < 2; ++ka) {
      const bf16* pr = &Pw[l16 * 64 + ka * 32];
      pf[ka].h[0] = *(const v8bf*)(pr + 8 * hl);
      pf[ka].h[1] = *(const v8bf*)(pr + 16 + 8 * hl);
    }
    // O += P V   (8 d-subtiles x 2 key-chunks)
#pragma unroll
    for (int nb2 = 0; nb2 < 8; ++nb2) {
#pragma unroll
      for (int ka = 0; ka < 2; ++ka) {
        Frag16 vf;
        const int d = nb2 * 16 + l16;
        vf.h[0] = *(const v8bf*)&Vt[d * 72 + ka * 32 + 8 * hl];
        vf.h[1] = *(const v8bf*)&Vt[d * 72 + ka * 32 + 16 + 8 * hl];
        acc[nb2] = wmma_bf16(pf[ka].v, vf.v, acc[nb2]);
      }
    }
  }
  // epilogue: normalize, write bf16 to [token, h*128+d]
#pragma unroll
  for (int nb2 = 0; nb2 < 8; ++nb2)
#pragma unroll
    for (int v = 0; v < 8; ++v) {
      const int qr = qrow0 + v + 8 * hl;
      const int d  = nb2 * 16 + l16;
      attn[((size_t)(b * S + qr)) * 4096 + h * 128 + d] = (bf16)(acc[nb2][v] / lrow[v]);
    }
}

// ---------------------------------------------------------------------------
// Host-side orchestration
// ---------------------------------------------------------------------------
extern "C" void kernel_launch(void* const* d_in, const int* in_sizes, int n_in,
                              void* d_out, int out_size, void* d_ws, size_t ws_size,
                              hipStream_t stream) {
  constexpr int M = 4096;           // B*S tokens
  const float* h_f   = (const float*)d_in[0];
  const float* Wq_f  = (const float*)d_in[1];
  const float* Wkr_f = (const float*)d_in[2];
  const float* Wdk_f = (const float*)d_in[3];
  const float* Wupk_f= (const float*)d_in[4];
  const float* Wupv_f= (const float*)d_in[5];
  const float* Wo_f  = (const float*)d_in[6];
  float* out = (float*)d_out;

  char* ws = (char*)d_ws;
  size_t off = 0;
  auto carve = [&](size_t bytes) -> char* {
    char* p = ws + off;
    off = (off + bytes + 255) & ~(size_t)255;
    return p;
  };
  bf16*  hb    = (bf16*)carve((size_t)M * 4096 * 2);
  bf16*  Wqb   = (bf16*)carve((size_t)4096 * 4096 * 2);
  bf16*  Wkrb  = (bf16*)carve((size_t)512 * 4096 * 2);
  bf16*  Wdkb  = (bf16*)carve((size_t)512 * 4096 * 2);
  bf16*  Wupkb = (bf16*)carve((size_t)512 * 512 * 2);
  bf16*  Wupvb = (bf16*)carve((size_t)1024 * 512 * 2);
  bf16*  Wob   = (bf16*)carve((size_t)4096 * 4096 * 2);
  float* qraw  = (float*)carve((size_t)M * 4096 * 4);
  float* kr0   = (float*)carve((size_t)M * 512 * 4);
  float* ckv   = (float*)carve((size_t)M * 512 * 4);
  bf16*  ckvb  = (bf16*)carve((size_t)M * 512 * 2);
  float* kcn   = (float*)carve((size_t)M * 512 * 4);
  float* vraw  = (float*)carve((size_t)M * 1024 * 4);
  bf16*  qb2   = (bf16*)carve((size_t)M * 4096 * 2);
  bf16*  kb2   = (bf16*)carve((size_t)M * 1024 * 2);
  bf16*  vb2   = (bf16*)carve((size_t)M * 1024 * 2);
  bf16*  attnb = (bf16*)carve((size_t)M * 4096 * 2);

  auto cvt = [&](const float* src, bf16* dst, int n) {
    cvt_f32_bf16<<<(n + 255) / 256, 256, 0, stream>>>(src, dst, n);
  };
  cvt(h_f,    hb,    M * 4096);
  cvt(Wq_f,   Wqb,   4096 * 4096);
  cvt(Wkr_f,  Wkrb,  512 * 4096);
  cvt(Wdk_f,  Wdkb,  512 * 4096);
  cvt(Wupk_f, Wupkb, 512 * 512);
  cvt(Wupv_f, Wupvb, 1024 * 512);
  cvt(Wo_f,   Wob,   4096 * 4096);

  auto gemm = [&](const bf16* A, const bf16* B, float* C, int m, int n, int k) {
    dim3 grid(m / BM, n / BN);
    gemm_bf16<<<grid, 256, 0, stream>>>(A, B, C, m, n, k);
  };
  gemm(hb, Wqb,  qraw, M, 4096, 4096);   // q projection
  gemm(hb, Wkrb, kr0,  M, 512,  4096);   // rope-k projection
  gemm(hb, Wdkb, ckv,  M, 512,  4096);   // low-rank kv
  cvt(ckv, ckvb, M * 512);
  gemm(ckvb, Wupkb, kcn,  M, 512,  512); // up-proj nope-k
  gemm(ckvb, Wupvb, vraw, M, 1024, 512); // up-proj v

  rope_q   <<<(M * 32 * 64) / 256, 256, 0, stream>>>(qraw, qb2);
  build_k  <<<(M * 8 * 64)  / 256, 256, 0, stream>>>(kr0, kcn, kb2);
  reshape_v<<<(M * 1024)    / 256, 256, 0, stream>>>(vraw, vb2);

  dim3 agrid(2048 / 64, 2 * 32);
  mla_attn<<<agrid, 128, 0, stream>>>(qb2, kb2, vb2, attnb);

  gemm(attnb, Wob, out, M, 4096, 4096);  // output projection -> f32 d_out
}
